// Model_82360292868732
// MI455X (gfx1250) — matrix operand order repaired and checked by reading, not compiled, not run on hardware
//
#include <hip/hip_runtime.h>
#include <math.h>

typedef __bf16 bf16_t;
typedef unsigned short u16;
typedef __attribute__((ext_vector_type(16))) __bf16 v16bf;
typedef __attribute__((ext_vector_type(8)))  __bf16 v8bf;
typedef __attribute__((ext_vector_type(8)))  float  v8f;
typedef __attribute__((ext_vector_type(4)))  int    v4i;

// ---------------- gfx1250 async global->LDS copy (guarded) ----------------

#if __has_builtin(__builtin_amdgcn_global_load_async_to_lds_b128) && \
    __has_builtin(__builtin_amdgcn_s_wait_asynccnt)
#define USE_ASYNC_LDS 1
#endif

__device__ __forceinline__ void cp16_g2l(u16* lds_dst, const u16* gsrc) {
#ifdef USE_ASYNC_LDS
    __builtin_amdgcn_global_load_async_to_lds_b128(
        (__attribute__((address_space(1))) v4i*)(gsrc),
        (__attribute__((address_space(3))) v4i*)(lds_dst), 0, 0);
#else
    *(uint4*)lds_dst = *(const uint4*)gsrc;
#endif
}

__device__ __forceinline__ void async_lds_fence() {
#ifdef USE_ASYNC_LDS
    __builtin_amdgcn_s_wait_asynccnt(0);
#endif
}

// ---------------- helpers ----------------

__device__ __forceinline__ u16 f2bf(float f) {
    unsigned int u = __float_as_uint(f);
    return (u16)((u + 0x7FFFu + ((u >> 16) & 1u)) >> 16);
}

__device__ __forceinline__ v16bf pack16(v8bf lo, v8bf hi) {
    v16bf r;
#pragma unroll
    for (int i = 0; i < 8; ++i) { r[i] = lo[i]; r[8 + i] = hi[i]; }
    return r;
}

// A fragment: 16x32 bf16 tile, A row-major (lda elements), A pre-offset to tile row 0.
// lanes 0-15: row=lane, K {k0..k0+7, k0+16..k0+23}; lanes 16-31: row=lane-16, K {+8, +24}
__device__ __forceinline__ v16bf load_a(const bf16_t* A, int lda, int k0, int lane) {
    const bf16_t* p = A + (size_t)(lane & 15) * lda + (k0 + ((lane >> 4) << 3));
    return pack16(*(const v8bf*)p, *(const v8bf*)(p + 16));
}

// B fragment: 32x16 bf16 tile from Bt stored N-major (K contiguous, ldk elements).
// lanes 0-15: col=lane, K {k0..k0+7, k0+16..k0+23}; lanes 16-31: col=lane-16, K {+8, +24}
__device__ __forceinline__ v16bf load_b(const bf16_t* Bt, int ldk, int k0, int lane) {
    const bf16_t* p = Bt + (size_t)(lane & 15) * ldk + (k0 + ((lane >> 4) << 3));
    return pack16(*(const v8bf*)p, *(const v8bf*)(p + 16));
}

__device__ __forceinline__ v8f wmma_bf16(v16bf a, v16bf b, v8f c) {
    return __builtin_amdgcn_wmma_f32_16x16x32_bf16(false, a, false, b, (short)0, c,
                                                   false, false);
}

__device__ __forceinline__ float gelu_exact(float y) {
    return 0.5f * y * (1.0f + erff(y * 0.70710678118654752f));
}

// ---------------- weight prep ----------------

// conv1_w (64,16,4,4) OIHW -> w1t[o][kh*64+kw*16+c]  (N=64 major, K=256 contiguous)
__global__ void k_prep_w1(const float* __restrict__ w, u16* __restrict__ w1t) {
    int i = blockIdx.x * 256 + threadIdx.x;
    if (i >= 64 * 256) return;
    int o = i >> 8, k = i & 255;
    int kh = k >> 6, kw = (k >> 4) & 3, c = k & 15;
    w1t[i] = f2bf(w[((o * 16 + c) * 4 + kh) * 4 + kw]);
}

// conv2_w (256,64,4,4) OIHW -> w2t[o][kh*256+kw*64+ci]  (N=256 major, K=1024 contiguous)
__global__ void k_prep_w2(const float* __restrict__ w, u16* __restrict__ w2t) {
    int i = blockIdx.x * 256 + threadIdx.x;
    if (i >= 256 * 1024) return;
    int o = i >> 10, k = i & 1023;
    int kh = k >> 8, kw = (k >> 6) & 3, ci = k & 63;
    w2t[i] = f2bf(w[((o * 64 + ci) * 4 + kh) * 4 + kw]);
}

// generic tiled transpose + f32->bf16 : in (R,C) -> out (C,R), batched on blockIdx.z
__global__ __launch_bounds__(128) void k_transpose_cvt(const float* __restrict__ in,
                                                       u16* __restrict__ out,
                                                       int R, int C) {
    __shared__ float tile[32][33];
    const float* src = in + (size_t)blockIdx.z * R * C;
    u16* dst = out + (size_t)blockIdx.z * R * C;
    int c0 = blockIdx.x * 32, r0 = blockIdx.y * 32;
    int tx = threadIdx.x & 31, ty = threadIdx.x >> 5;
    for (int j = ty; j < 32; j += 4)
        tile[j][tx] = src[(size_t)(r0 + j) * C + (c0 + tx)];
    __syncthreads();
    for (int j = ty; j < 32; j += 4)
        dst[(size_t)(c0 + j) * R + (r0 + tx)] = f2bf(tile[tx][j]);
}

// ---------------- stage 1: conv1 (implicit GEMM) + LN + GeLU ----------------
// x: (1024, 64, 64, 16) f32.  M=262144 pixels, K=256, N=64.
// act1 out: (1024, 16, 16, 64) bf16.
__global__ __launch_bounds__(128) void k_conv1(const float* __restrict__ x,
                                               const u16* __restrict__ w1t,
                                               const float* __restrict__ cb,
                                               const float* __restrict__ lg,
                                               const float* __restrict__ lb,
                                               u16* __restrict__ act1) {
    __shared__ alignas(16) u16 sA[64 * 264];   // 64 pixels x (256 K + pad)
    __shared__ float sOut[64 * 65];            // 64 pixels x 64 ch (+pad)
    const int tid = threadIdx.x, lane = tid & 31, wave = tid >> 5;
    const int pix0 = blockIdx.x * 64;

    // cooperative load + f32->bf16: 256 segments of 64 contiguous floats
    for (int s = tid; s < 256; s += 128) {
        int pl = s >> 2, kh = s & 3;
        int p = pix0 + pl;
        int bt = p >> 8, ph = (p >> 4) & 15, pw = p & 15;
        const float* src = x + (size_t)bt * 65536 + (size_t)(ph * 4 + kh) * 1024 + pw * 64;
        if (bt + 1 < 1024) __builtin_prefetch(src + 65536, 0, 1);
        u16* dst = sA + pl * 264 + kh * 64;
#pragma unroll
        for (int j = 0; j < 16; ++j) {
            float4 v = ((const float4*)src)[j];
            uint2 pk;
            pk.x = (unsigned)f2bf(v.x) | ((unsigned)f2bf(v.y) << 16);
            pk.y = (unsigned)f2bf(v.z) | ((unsigned)f2bf(v.w) << 16);
            ((uint2*)dst)[j] = pk;
        }
    }
    __syncthreads();

    // GEMM: wave w owns channels [w*16, w*16+16); 4 M-subtiles of 16 pixels
    const bf16_t* Bt = (const bf16_t*)w1t + (size_t)wave * 16 * 256;
    for (int msub = 0; msub < 4; ++msub) {
        v8f acc = {};
        const bf16_t* At = (const bf16_t*)sA + (size_t)msub * 16 * 264;
#pragma unroll
        for (int kc = 0; kc < 8; ++kc) {
            v16bf a = load_a(At, 264, kc * 32, lane);
            v16bf b = load_b(Bt, 256, kc * 32, lane);
            acc = wmma_bf16(a, b, acc);
        }
        int mrow = msub * 16 + (lane >> 4) * 8;
        int ncol = wave * 16 + (lane & 15);
#pragma unroll
        for (int r = 0; r < 8; ++r) sOut[(mrow + r) * 65 + ncol] = acc[r];
    }
    __syncthreads();

    // LN over 64 channels + GeLU, store bf16
    if (tid < 64) {
        float mean = 0.f;
        for (int i = 0; i < 64; ++i) mean += sOut[tid * 65 + i] + cb[i];
        mean *= (1.f / 64.f);
        float var = 0.f;
        for (int i = 0; i < 64; ++i) {
            float d = sOut[tid * 65 + i] + cb[i] - mean;
            var += d * d;
        }
        var *= (1.f / 64.f);
        float inv = rsqrtf(var + 1e-5f);
        u16* dst = act1 + (size_t)(pix0 + tid) * 64;
        for (int i = 0; i < 64; ++i) {
            float y = (sOut[tid * 65 + i] + cb[i] - mean) * inv * lg[i] + lb[i];
            dst[i] = f2bf(gelu_exact(y));
        }
    }
}

// ---------------- stage 2: conv2 (implicit GEMM) + LN + GeLU ----------------
// act1: (1024, 16, 16, 64) bf16.  M=16384 pixels, K=1024, N=256.
// 64-pixel M-tile: B fragments loaded once per K-step and reused by 4 M-subtiles
// (4x less L2 traffic on w2t).  A staged in LDS one kh-slab (K=256) at a time.
// act2 out: (1024, 4, 4, 256) bf16 == fc input rows of 4096.
__global__ __launch_bounds__(128) void k_conv2(const u16* __restrict__ act1,
                                               const u16* __restrict__ w2t,
                                               const float* __restrict__ cb,
                                               const float* __restrict__ lg,
                                               const float* __restrict__ lb,
                                               u16* __restrict__ act2) {
    __shared__ alignas(16) u16 sA[64 * 264];   // 64 pixels x one 256-K slab (+pad)
    __shared__ float sOut[64 * 257];           // 64 pixels x 256 ch (+pad)
    const int tid = threadIdx.x, lane = tid & 31, wave = tid >> 5;
    const int pix0 = blockIdx.x * 64;

    v8f acc[4][4] = {};                        // [msub][ntile j]
    for (int kh = 0; kh < 4; ++kh) {
        // stage one kh slab: per pixel 256 contiguous bf16 (async copy to LDS)
        for (int c = tid; c < 2048; c += 128) {
            int pl = c >> 5, off = c & 31;
            int p = pix0 + pl;
            int bt = p >> 4, ph = (p >> 2) & 3, pw = p & 3;
            const u16* src = act1 + (size_t)bt * 16384 + (size_t)(ph * 4 + kh) * 1024 +
                             pw * 256 + off * 8;
            cp16_g2l(sA + pl * 264 + off * 8, src);
        }
        async_lds_fence();
        __syncthreads();

        for (int kc = 0; kc < 8; ++kc) {
            v16bf bfr[4];
#pragma unroll
            for (int j = 0; j < 4; ++j) {
                const bf16_t* Bt = (const bf16_t*)w2t + (size_t)(wave * 4 + j) * 16 * 1024;
                bfr[j] = load_b(Bt, 1024, kh * 256 + kc * 32, lane);
            }
#pragma unroll
            for (int msub = 0; msub < 4; ++msub) {
                v16bf a = load_a((const bf16_t*)sA + (size_t)msub * 16 * 264, 264,
                                 kc * 32, lane);
#pragma unroll
                for (int j = 0; j < 4; ++j)
                    acc[msub][j] = wmma_bf16(a, bfr[j], acc[msub][j]);
            }
        }
        __syncthreads();
    }

    int mr = (lane >> 4) * 8;
    int ncol = lane & 15;
#pragma unroll
    for (int msub = 0; msub < 4; ++msub)
#pragma unroll
        for (int j = 0; j < 4; ++j) {
            int n = (wave * 4 + j) * 16 + ncol;
#pragma unroll
            for (int r = 0; r < 8; ++r)
                sOut[(msub * 16 + mr + r) * 257 + n] = acc[msub][j][r];
        }
    __syncthreads();

    if (tid < 64) {
        float mean = 0.f;
        for (int i = 0; i < 256; ++i) mean += sOut[tid * 257 + i] + cb[i];
        mean *= (1.f / 256.f);
        float var = 0.f;
        for (int i = 0; i < 256; ++i) {
            float d = sOut[tid * 257 + i] + cb[i] - mean;
            var += d * d;
        }
        var *= (1.f / 256.f);
        float inv = rsqrtf(var + 1e-5f);
        u16* dst = act2 + (size_t)(pix0 + tid) * 256;
        for (int i = 0; i < 256; ++i) {
            float y = (sOut[tid * 257 + i] + cb[i] - mean) * inv * lg[i] + lb[i];
            dst[i] = f2bf(gelu_exact(y));
        }
    }
}

// ---------------- stage 3: fc GEMM (1024x4096 @ 4096x512 + b) ----------------
// 64-row M-tile; B fragment loaded once per K-step, reused by 4 M-subtiles.
__global__ __launch_bounds__(128) void k_fc(const u16* __restrict__ act2,
                                            const u16* __restrict__ fct,
                                            const float* __restrict__ fb,
                                            float* __restrict__ embf,
                                            u16* __restrict__ embh) {
    __shared__ alignas(16) u16 sA[64 * 264];
    const int tid = threadIdx.x, lane = tid & 31, wave = tid >> 5;
    const int m0 = blockIdx.x * 64;
    const int nBase = blockIdx.y * 64 + wave * 16;
    const bf16_t* Bt = (const bf16_t*)fct + (size_t)nBase * 4096;
    v8f acc[4] = {};
    for (int kc = 0; kc < 16; ++kc) {
        for (int c = tid; c < 2048; c += 128) {
            int r = c >> 5, off = c & 31;
            cp16_g2l(sA + r * 264 + off * 8,
                     act2 + (size_t)(m0 + r) * 4096 + kc * 256 + off * 8);
        }
        async_lds_fence();
        __syncthreads();
#pragma unroll
        for (int kk = 0; kk < 8; ++kk) {
            v16bf b = load_b(Bt, 4096, kc * 256 + kk * 32, lane);
#pragma unroll
            for (int msub = 0; msub < 4; ++msub) {
                v16bf a = load_a((const bf16_t*)sA + (size_t)msub * 16 * 264, 264,
                                 kk * 32, lane);
                acc[msub] = wmma_bf16(a, b, acc[msub]);
            }
        }
        __syncthreads();
    }
    int n = nBase + (lane & 15);
    float bias = fb[n];
#pragma unroll
    for (int msub = 0; msub < 4; ++msub) {
        int mr = m0 + msub * 16 + (lane >> 4) * 8;
#pragma unroll
        for (int r = 0; r < 8; ++r) {
            float v = acc[msub][r] + bias;
            embf[(size_t)(mr + r) * 512 + n] = v;
            embh[(size_t)(mr + r) * 512 + n] = f2bf(v);
        }
    }
}

// ---------------- stage 4: Fourier top-2 gating ----------------
__global__ __launch_bounds__(128) void k_gate(const float* __restrict__ emb,
                                              const float* __restrict__ w_gate,
                                              float* __restrict__ gates) {
    __shared__ float cs[128], sn[128];
    __shared__ float red[6 * 128];
    const int b = blockIdx.x, tid = threadIdx.x;
    {
        int f = tid >> 4, t = tid & 15;
        float ang = -6.283185307179586f * (float)((f + 1) * t) / 16.0f;
        cs[tid] = cosf(ang);
        sn[tid] = sinf(ang);
    }
    __syncthreads();
    float acc[6] = {0.f, 0.f, 0.f, 0.f, 0.f, 0.f};
    for (int d = tid; d < 512; d += 128) {
        float xv[16];
#pragma unroll
        for (int t = 0; t < 16; ++t) xv[t] = emb[((size_t)b * 16 + t) * 512 + d];
#pragma unroll
        for (int f = 0; f < 8; ++f) {
            float re = 0.f, im = 0.f;
#pragma unroll
            for (int t = 0; t < 16; ++t) {
                re += xv[t] * cs[f * 16 + t];
                im += xv[t] * sn[f * 16 + t];
            }
            float amp = 0.25f * sqrtf(re * re + im * im);  // ortho norm 1/sqrt(16)
#pragma unroll
            for (int e = 0; e < 6; ++e) acc[e] += amp * w_gate[f * 6 + e];
        }
    }
    for (int e = 0; e < 6; ++e) red[e * 128 + tid] = acc[e];
    __syncthreads();
    for (int s = 64; s > 0; s >>= 1) {
        if (tid < s)
            for (int e = 0; e < 6; ++e) red[e * 128 + tid] += red[e * 128 + tid + s];
        __syncthreads();
    }
    if (tid == 0) {
        float w[6];
        for (int e = 0; e < 6; ++e) w[e] = red[e * 128] * (1.f / 512.f);
        int i0 = 0;
        for (int e = 1; e < 6; ++e) if (w[e] > w[i0]) i0 = e;
        int i1 = (i0 == 0) ? 1 : 0;
        for (int e = 0; e < 6; ++e) if (e != i0 && w[e] > w[i1]) i1 = e;
        float g1 = expf(w[i1] - w[i0]);
        float s = 1.0f + g1;
        float out[6] = {0.f, 0.f, 0.f, 0.f, 0.f, 0.f};
        out[i0] = 1.0f / s;
        out[i1] = g1 / s;
        for (int e = 0; e < 6; ++e) gates[b * 6 + e] = out[e];
    }
}

// ---------------- stage 5: experts + gated exp/sum/log combine ----------------
__global__ __launch_bounds__(128) void k_experts(const u16* __restrict__ embh,
                                                 const u16* __restrict__ ewt,
                                                 const float* __restrict__ eb,
                                                 const float* __restrict__ gates,
                                                 float* __restrict__ out) {
    const int tid = threadIdx.x, lane = tid & 31, wave = tid >> 5;
    const int b = blockIdx.x;
    const int nBase = blockIdx.y * 64 + wave * 16;
    const bf16_t* At = (const bf16_t*)embh + (size_t)b * 16 * 512;
    const int n = nBase + (lane & 15);
    v8f comb = {};
    for (int e = 0; e < 6; ++e) {
        const bf16_t* Bt = (const bf16_t*)ewt + (size_t)e * 512 * 512 + (size_t)nBase * 512;
        v8f acc = {};
#pragma unroll
        for (int kc = 0; kc < 16; ++kc) {
            v16bf a = load_a(At, 512, kc * 32, lane);
            v16bf bf = load_b(Bt, 512, kc * 32, lane);
            acc = wmma_bf16(a, bf, acc);
        }
        float g = gates[b * 6 + e];          // wave-uniform (gate==0 contributes 0)
        float bias = eb[e * 512 + n];
#pragma unroll
        for (int r = 0; r < 8; ++r) comb[r] += g * expf(acc[r] + bias);
    }
    int mr = b * 16 + (lane >> 4) * 8;
#pragma unroll
    for (int r = 0; r < 8; ++r) {
        float v = comb[r];
        v = (v == 0.f) ? 2.220446049250313e-16f : v;
        out[(size_t)(mr + r) * 512 + n] = logf(v);
    }
}

// ---------------- launch ----------------

extern "C" void kernel_launch(void* const* d_in, const int* in_sizes, int n_in,
                              void* d_out, int out_size, void* d_ws, size_t ws_size,
                              hipStream_t stream) {
    const float* x        = (const float*)d_in[0];
    const float* conv1_w  = (const float*)d_in[1];
    const float* conv1_b  = (const float*)d_in[2];
    const float* ln1_g    = (const float*)d_in[3];
    const float* ln1_b    = (const float*)d_in[4];
    const float* conv2_w  = (const float*)d_in[5];
    const float* conv2_b  = (const float*)d_in[6];
    const float* ln2_g    = (const float*)d_in[7];
    const float* ln2_b    = (const float*)d_in[8];
    const float* fc_w     = (const float*)d_in[9];
    const float* fc_b     = (const float*)d_in[10];
    const float* w_gate   = (const float*)d_in[11];
    const float* expert_w = (const float*)d_in[12];
    const float* expert_b = (const float*)d_in[13];
    // d_in[14] = top_k (always 2 per reference setup)

    char* ws = (char*)d_ws;
    u16*   w1t   = (u16*)(ws + 0);          //  64x256  bf16
    u16*   w2t   = (u16*)(ws + 32768);      // 256x1024 bf16
    u16*   fct   = (u16*)(ws + 557056);     // 512x4096 bf16
    u16*   ewt   = (u16*)(ws + 4751360);    // 6x512x512 bf16 (transposed)
    u16*   act1  = (u16*)(ws + 7897088);    // 262144x64 bf16
    u16*   act2  = (u16*)(ws + 41451520);   // 16384x256 bf16
    float* embf  = (float*)(ws + 49840128); // 1024x512 f32
    u16*   embh  = (u16*)(ws + 51937280);   // 1024x512 bf16
    float* gates = (float*)(ws + 52985856); // 64x6 f32

    // weight prep
    k_prep_w1<<<64, 256, 0, stream>>>(conv1_w, w1t);
    k_prep_w2<<<1024, 256, 0, stream>>>(conv2_w, w2t);
    k_transpose_cvt<<<dim3(16, 128, 1), 128, 0, stream>>>(fc_w, fct, 4096, 512);
    k_transpose_cvt<<<dim3(16, 16, 6), 128, 0, stream>>>(expert_w, ewt, 512, 512);

    // pipeline
    k_conv1<<<4096, 128, 0, stream>>>(x, w1t, conv1_b, ln1_g, ln1_b, act1);
    k_conv2<<<256, 128, 0, stream>>>(act1, w2t, conv2_b, ln2_g, ln2_b, act2);
    k_fc<<<dim3(16, 8), 128, 0, stream>>>(act2, fct, fc_b, embf, embh);
    k_gate<<<64, 128, 0, stream>>>(embf, w_gate, gates);
    k_experts<<<dim3(64, 8), 128, 0, stream>>>(embh, ewt, expert_b, gates,
                                               (float*)d_out);
}
